// UNetKPCMMedium_50379966382788
// MI455X (gfx1250) — matrix operand order repaired and checked by reading, not compiled, not run on hardware
//
#include <hip/hip_runtime.h>
#include <hip/hip_bf16.h>

typedef _Float16 f16;
typedef __attribute__((ext_vector_type(16))) _Float16 v16h;
typedef __attribute__((ext_vector_type(8)))  _Float16 v8h;
typedef __attribute__((ext_vector_type(8)))  float    v8f;
typedef __attribute__((ext_vector_type(4)))  unsigned int u32x4;
typedef __attribute__((ext_vector_type(8)))  int i32x8;
typedef __attribute__((ext_vector_type(4)))  int i32x4;

union V16 { v16h v; v8h h[2]; };

#if defined(__has_builtin)
#if __has_builtin(__builtin_amdgcn_tensor_load_to_lds) && __has_builtin(__builtin_amdgcn_s_wait_tensorcnt)
#define USE_TDM 1
#endif
#endif
#ifndef USE_TDM
#define USE_TDM 0
#endif

// ---------------------------------------------------------------------------
// WMMA fragment index helpers (16-bit A 16x32 / B 32x16 layouts, wave32).
// A: lane = m + 16*g; element e -> K = e + 16*g (16 contiguous K per half)
//    => the two 8-element halves are K-contiguous: vector-loadable (2 x b128).
// B: lane = n + 16*g; element e -> K = e + 16*g (16 contiguous K per half).
// C/D: lane = n + 16*g; VGPR r -> M = r + 8*g.
// ---------------------------------------------------------------------------
__device__ __forceinline__ int a_k_of(int g, int e) {
  return e + 16 * g;
}

// ---------------------------------------------------------------------------
// Generic conv (used only for layer 1: Cin=3, K=27). Guarded scalar loads.
// Weights in flat OIHW [Cout][Cin*9].
// ---------------------------------------------------------------------------
__global__ void conv_wmma_generic(const f16* __restrict__ x, const f16* __restrict__ w,
                                  const float* __restrict__ bias, f16* __restrict__ y,
                                  int Cin, int Cout, int H, int W, int relu, int nTilesN)
{
  const int wave = threadIdx.x >> 5;
  const int lane = threadIdx.x & 31;
  const int tile = blockIdx.x * (blockDim.x >> 5) + wave;
  const int nTilesM = (Cout + 15) >> 4;
  if (tile >= nTilesM * nTilesN) return;            // wave-uniform exit
  const int mt = tile / nTilesN, nt = tile - mt * nTilesN;
  const int m_base = mt << 4, n_base = nt << 4;
  const int g = lane >> 4, l = lane & 15;
  const int HW = H * W;
  const int K = Cin * 9;
  const int n = n_base + l;
  const int oh = n / W, ow = n - oh * W;
  const int m = m_base + l;

  v8f acc = {};
  for (int k0 = 0; k0 < K; k0 += 32) {
    v16h a, b;
#pragma unroll
    for (int e = 0; e < 16; ++e) {
      const int ka = k0 + a_k_of(g, e);
      a[e] = (ka < K && m < Cout) ? w[(size_t)m * K + ka] : (f16)0.f;
      const int kb = k0 + g * 16 + e;
      f16 vb = (f16)0.f;
      if (kb < K) {
        const int ci = kb / 9, r = kb - ci * 9;
        const int kh = r / 3, kw = r - kh * 3;
        const int ih = oh + kh - 1, iw = ow + kw - 1;
        if (ih >= 0 && ih < H && iw >= 0 && iw < W)
          vb = x[(size_t)ci * HW + ih * W + iw];
      }
      b[e] = vb;
    }
    acc = __builtin_amdgcn_wmma_f32_16x16x32_f16(false, a, false, b,
                                                 (short)0, acc, false, false);
  }
#pragma unroll
  for (int r = 0; r < 8; ++r) {
    const int mm = m_base + r + 8 * g;
    if (mm < Cout) {
      float v = acc[r] + bias[mm];
      if (relu) v = v > 0.f ? v : 0.f;
      y[(size_t)mm * HW + n] = (f16)v;
    }
  }
}

// ---------------------------------------------------------------------------
// Tiled 3x3 conv, pad=1, Cin in {64,128,256}, Cout multiple of 16.
// Block = 256 thr (8 waves) = one 16-pixel row strip.
// Halo staging: the Cin x 3 x 18 input tile goes to LDS either via the Tensor
// Data Mover (interior blocks: 3D descriptor, x-fast = 18 cols, then 3 rows
// stride W, then Cin channels stride H*W -> lands in [ci][3][18] order) or a
// cooperative zero-padded loop at image borders. Weights are tap-major
// [Cout][t][ci]: tap index uniform per fragment half, A = two b128 loads.
// ---------------------------------------------------------------------------
__global__ void conv3_tiled_wmma(const f16* __restrict__ x, const f16* __restrict__ w,
                                 const float* __restrict__ bias, f16* __restrict__ y,
                                 int Cin, int cinShift, int Cout, int H, int W, int relu)
{
  extern __shared__ char dynsmem[];
  f16* lds = (f16*)dynsmem;                         // [Cin][3][18]

  const int tid  = threadIdx.x;
  const int wave = tid >> 5;
  const int lane = tid & 31;
  const int g = lane >> 4, l = lane & 15;
  const int HW = H * W;
  const int n_base = blockIdx.x << 4;
  const int oh = n_base / W, ow_base = n_base - oh * W;

  bool staged = false;
#if USE_TDM
  const bool interior = (oh >= 1) && (oh < H - 1) && (ow_base >= 1) && (ow_base + 17 <= W);
  if (interior) {                                    // block-uniform
    if (tid < 32) {                                  // wave 0 issues the DMA
      const unsigned lds_addr = (unsigned)(size_t)(void*)lds;
      const unsigned long long ga =
          (unsigned long long)(size_t)(x + (size_t)(oh - 1) * W + (ow_base - 1));
      u32x4 g0 = (u32x4)0u;
      g0[0] = 1u;                                    // count=1, user D#
      g0[1] = lds_addr;                              // LDS dest (bytes)
      g0[2] = (unsigned)(ga & 0xffffffffu);
      g0[3] = (unsigned)((ga >> 32) & 0x01ffffffu) | (2u << 30);   // type=2
      i32x8 g1 = (i32x8)0;
      g1[0] = (1 << 16);                             // data_size = 2 bytes
      g1[1] = (18 << 16);                            // tensor_dim0 = 18
      g1[2] = (3 << 16);                             // tensor_dim1 = 3
      g1[3] = (18 << 16);                            // tile_dim0 = 18
      g1[4] = 3 | (Cin << 16);                       // tile_dim1=3, tile_dim2=Cin
      g1[5] = W;                                     // dim0 stride (rows) = W
      g1[6] = (int)((HW & 0xffff) << 16);            // dim1 stride (chan) lo16
      g1[7] = (int)(HW >> 16);                       //   .. bits 47:16
      i32x4 g2 = (i32x4)0;
      g2[0] = Cin;                                   // tensor_dim2 = Cin
      i32x4 g3 = (i32x4)0;
      i32x8 g4 = (i32x8)0;
      __builtin_amdgcn_tensor_load_to_lds(g0, g1, g2, g3, g4, 0);
      __builtin_amdgcn_s_wait_tensorcnt(0);
    }
    staged = true;
  }
#endif
  if (!staged) {                                     // border fallback, zero-padded
    const int total = Cin * 54;
    for (int idx = tid; idx < total; idx += blockDim.x) {
      const int ci = idx / 54, rem = idx - ci * 54;
      const int r = rem / 18, c = rem - r * 18;
      const int ih = oh + r - 1, iw = ow_base + c - 1;
      f16 v = (f16)0.f;
      if (ih >= 0 && ih < H && iw >= 0 && iw < W)
        v = x[(size_t)ci * HW + ih * W + iw];
      lds[idx] = v;
    }
  }
  __syncthreads();

  const int K = Cin * 9;
  const int nTilesM = Cout >> 4;
  const int cinMask = Cin - 1;
  const int n = n_base + l;

  for (int mt = wave; mt < nTilesM; mt += (int)(blockDim.x >> 5)) {
    const int m_base = mt << 4;
    const int m = m_base + l;
    const f16* wrow = w + (size_t)m * K;

    v8f acc = {};
    for (int k0 = 0; k0 < K; k0 += 32) {             // K multiple of 32
      __builtin_prefetch((const void*)(wrow + k0 + 128), 0, 1);
      V16 a;                                         // 2 x global_load_b128
      a.h[0] = *(const v8h*)(wrow + k0 + 16 * g);
      a.h[1] = *(const v8h*)(wrow + k0 + 8 + 16 * g);
      const int kbh = k0 + 16 * g;
      const int t = kbh >> cinShift;                 // tap, uniform over e
      const int ci_base = kbh & cinMask;
      const int kh = (t >= 6) ? 2 : ((t >= 3) ? 1 : 0);
      const int kw = t - 3 * kh;
      const f16* bp = lds + (size_t)ci_base * 54 + kh * 18 + l + kw;
      v16h b;
#pragma unroll
      for (int e = 0; e < 16; ++e) b[e] = bp[e * 54];
      acc = __builtin_amdgcn_wmma_f32_16x16x32_f16(false, a.v, false, b,
                                                   (short)0, acc, false, false);
    }
#pragma unroll
    for (int r = 0; r < 8; ++r) {
      const int mm = m_base + r + 8 * g;
      float v = acc[r] + bias[mm];
      if (relu) v = v > 0.f ? v : 0.f;
      y[(size_t)mm * HW + n] = (f16)v;
    }
  }
}

// ---------------------------------------------------------------------------
// ConvTranspose2d(k=2,s=2) as 4 parity GEMMs. Input tile [Cin][16] staged to
// LDS via TDM (2D descriptor: 16 contiguous pixels, Cin rows stride Hin*Win;
// no OOB possible so TDM is unconditional), manual fallback otherwise.
// wt pre-packed [par][Cout][Cin] f16.
// ---------------------------------------------------------------------------
__global__ void tconv_tiled_wmma(const f16* __restrict__ x, const f16* __restrict__ wt,
                                 const float* __restrict__ bias, f16* __restrict__ y,
                                 int Cin, int Cout, int Hin, int Win)
{
  extern __shared__ char dynsmem[];
  f16* xs = (f16*)dynsmem;                           // [Cin][16]

  const int tid  = threadIdx.x;
  const int wave = tid >> 5;
  const int lane = tid & 31;
  const int g = lane >> 4, l = lane & 15;
  const int HWin = Hin * Win;
  const int n_base = blockIdx.x << 4;

#if USE_TDM
  if (tid < 32) {                                    // wave 0 issues the DMA
    const unsigned lds_addr = (unsigned)(size_t)(void*)xs;
    const unsigned long long ga = (unsigned long long)(size_t)(x + n_base);
    u32x4 g0 = (u32x4)0u;
    g0[0] = 1u;
    g0[1] = lds_addr;
    g0[2] = (unsigned)(ga & 0xffffffffu);
    g0[3] = (unsigned)((ga >> 32) & 0x01ffffffu) | (2u << 30);
    i32x8 g1 = (i32x8)0;
    g1[0] = (1 << 16);                               // data_size = 2 bytes
    g1[1] = (16 << 16);                              // tensor_dim0 = 16
    g1[2] = (Cin << 16);                             // tensor_dim1 = Cin
    g1[3] = (16 << 16);                              // tile_dim0 = 16
    g1[4] = Cin;                                     // tile_dim1 = Cin
    g1[5] = HWin;                                    // dim0 stride = Hin*Win
    i32x4 g2 = (i32x4)0;
    i32x4 g3 = (i32x4)0;
    i32x8 g4 = (i32x8)0;
    __builtin_amdgcn_tensor_load_to_lds(g0, g1, g2, g3, g4, 0);
    __builtin_amdgcn_s_wait_tensorcnt(0);
  }
#else
  for (int idx = tid; idx < Cin * 16; idx += blockDim.x) {
    const int ci = idx >> 4, nl = idx & 15;          // coalesced over nl
    xs[ci * 16 + nl] = x[(size_t)ci * HWin + n_base + nl];
  }
#endif
  __syncthreads();

  const int n = n_base + l;
  const int h = n / Win, w2 = n - h * Win;
  const int Wout = Win << 1;
  const int HWout = HWin << 2;
  const int nTilesM = Cout >> 4;

  for (int tile = wave; tile < 4 * nTilesM; tile += (int)(blockDim.x >> 5)) {
    const int par = tile & 3, mt = tile >> 2;
    const int m_base = mt << 4;
    const f16* wrow = wt + ((size_t)par * Cout + m_base + l) * Cin;

    v8f acc = {};
    for (int k0 = 0; k0 < Cin; k0 += 32) {           // Cin = 128 or 256
      V16 a;
      a.h[0] = *(const v8h*)(wrow + k0 + 16 * g);
      a.h[1] = *(const v8h*)(wrow + k0 + 8 + 16 * g);
      const f16* bp = xs + (size_t)(k0 + 16 * g) * 16 + l;
      v16h b;
#pragma unroll
      for (int e = 0; e < 16; ++e) b[e] = bp[e * 16];
      acc = __builtin_amdgcn_wmma_f32_16x16x32_f16(false, a.v, false, b,
                                                   (short)0, acc, false, false);
    }
    const int dh = par >> 1, dw = par & 1;
    const int op = ((h << 1) + dh) * Wout + ((w2 << 1) + dw);
#pragma unroll
    for (int r = 0; r < 8; ++r) {
      const int mm = m_base + r + 8 * g;
      y[(size_t)mm * HWout + op] = (f16)(acc[r] + bias[mm]);
    }
  }
}

// ---------------------------------------------------------------------------
// Fused KPN head: 1x1 conv (64 -> 441 taps) via WMMA + normalization + 21x21
// weighted patch sum. raw (441x65536 f32 ~115 MB) never hits memory: the
// 441x16 tap weights for this 16-pixel tile live in LDS (28 KB of 320 KB/WGP).
// clean_pred = noisy + (out - noisy) = out.
// ---------------------------------------------------------------------------
__global__ void kpn_wmma_kernel(const f16* __restrict__ d1, const f16* __restrict__ wo,
                                const float* __restrict__ ob, const float* __restrict__ noisy,
                                float* __restrict__ out)
{
  __shared__ float wt[448 * 16];
  __shared__ float inv_dnm[16];
  const int lane = threadIdx.x;
  const int g = lane >> 4, l = lane & 15;
  const int n_base = blockIdx.x << 4;
  const int HW = 256 * 256;

  // B fragments for K=64 (two k-steps), loaded once, reused for 28 M tiles.
  v16h b0, b1;
#pragma unroll
  for (int e = 0; e < 16; ++e) {
    const int kb = g * 16 + e;
    b0[e] = d1[(size_t)kb * HW + n_base + l];
    b1[e] = d1[(size_t)(kb + 32) * HW + n_base + l];
  }

  for (int mtile = 0; mtile < 28; ++mtile) {         // 28*16 = 448 >= 441 taps
    const int m = (mtile << 4) + l;
    V16 a0, a1;
    if (m < 441) {
      const f16* wr = wo + m * 64;
      a0.h[0] = *(const v8h*)(wr + 16 * g);
      a0.h[1] = *(const v8h*)(wr + 8 + 16 * g);
      a1.h[0] = *(const v8h*)(wr + 32 + 16 * g);
      a1.h[1] = *(const v8h*)(wr + 40 + 16 * g);
    } else {
      a0.v = (v16h)(f16)0.f;  a1.v = (v16h)(f16)0.f;
    }
    v8f acc = {};
    acc = __builtin_amdgcn_wmma_f32_16x16x32_f16(false, a0.v, false, b0, (short)0, acc, false, false);
    acc = __builtin_amdgcn_wmma_f32_16x16x32_f16(false, a1.v, false, b1, (short)0, acc, false, false);
#pragma unroll
    for (int r = 0; r < 8; ++r) {
      const int mm = (mtile << 4) + r + 8 * g;
      wt[mm * 16 + l] = acc[r] + (mm < 441 ? ob[mm] : 0.f);
    }
  }
  __syncthreads();

  if (lane < 16) {                                    // denom = max(sum, 1e-6)
    float s = 0.f;
    for (int t = 0; t < 441; ++t) s += wt[t * 16 + lane];
    inv_dnm[lane] = 1.f / (s > 1e-6f ? s : 1e-6f);
  }
  __syncthreads();

  // 48 tasks = 3 channels x 16 pixels; 441-tap weighted patch sum each.
  for (int task = lane; task < 48; task += 32) {
    const int c = task >> 4, j = task & 15;
    const int p = n_base + j;
    const int y0 = p >> 8, x0 = p & 255;
    const float inv = inv_dnm[j];
    float s = 0.f;
    for (int t = 0; t < 441; ++t) {
      const int dy = t / 21 - 10, dx = t % 21 - 10;
      const int yy = y0 + dy, xx = x0 + dx;
      const float v = (yy >= 0 && yy < 256 && xx >= 0 && xx < 256)
                          ? noisy[c * HW + (yy << 8) + xx] : 0.f;
      s += wt[t * 16 + j] * inv * v;
    }
    out[c * HW + p] = s;
  }
}

// ---------------------------------------------------------------------------
// Helpers: f32->f16 convert, f16 copy, maxpool 2x2, weight repacks.
// ---------------------------------------------------------------------------
__global__ void cvt_f32_f16_kernel(const float* __restrict__ s, f16* __restrict__ d, int n) {
  for (int i = blockIdx.x * blockDim.x + threadIdx.x; i < n; i += gridDim.x * blockDim.x)
    d[i] = (f16)s[i];
}

__global__ void copy_f16_kernel(const f16* __restrict__ s, f16* __restrict__ d, int n) {
  for (int i = blockIdx.x * blockDim.x + threadIdx.x; i < n; i += gridDim.x * blockDim.x)
    d[i] = s[i];
}

__global__ void maxpool2_kernel(const f16* __restrict__ x, f16* __restrict__ y,
                                int C, int Hin, int Win) {
  const int Hout = Hin >> 1, Wout = Win >> 1;
  const int n = C * Hout * Wout;
  for (int i = blockIdx.x * blockDim.x + threadIdx.x; i < n; i += gridDim.x * blockDim.x) {
    const int c = i / (Hout * Wout);
    const int r = i - c * Hout * Wout;
    const int h = r / Wout, w = r - h * Wout;
    const f16* p = x + (size_t)c * Hin * Win + (h << 1) * Win + (w << 1);
    f16 a = p[0], b = p[1], c2 = p[Win], d = p[Win + 1];
    f16 m0 = a > b ? a : b;
    f16 m1 = c2 > d ? c2 : d;
    y[i] = m0 > m1 ? m0 : m1;
  }
}

// OIHW f32 -> tap-major [co][t][ci] f16 (k = t*Cin + ci).
__global__ void repack_conv_tapmajor(const float* __restrict__ src, f16* __restrict__ dst,
                                     int Cin, int Cout) {
  const int n = Cout * 9 * Cin;
  for (int i = blockIdx.x * blockDim.x + threadIdx.x; i < n; i += gridDim.x * blockDim.x) {
    const int co = i / (9 * Cin);
    const int rem = i - co * 9 * Cin;
    const int t = rem / Cin, ci = rem - t * Cin;
    dst[i] = (f16)src[((size_t)co * Cin + ci) * 9 + t];
  }
}

// ConvTranspose weights (in,out,kh,kw) f32 -> [par][out][in] f16.
__global__ void repack_tconv_kernel(const float* __restrict__ src, f16* __restrict__ dst,
                                    int Cin, int Cout) {
  const int n = Cin * Cout * 4;
  for (int i = blockIdx.x * blockDim.x + threadIdx.x; i < n; i += gridDim.x * blockDim.x) {
    const int par = i / (Cout * Cin);
    const int rem = i - par * Cout * Cin;
    const int co = rem / Cin, ci = rem - co * Cin;
    dst[i] = (f16)src[(size_t)(ci * Cout + co) * 4 + par];
  }
}

// ---------------------------------------------------------------------------
// Host orchestration
// ---------------------------------------------------------------------------
static inline int gs_blocks(long n) {
  long b = (n + 255) / 256;
  return (int)(b > 4096 ? 4096 : (b < 1 ? 1 : b));
}

extern "C" void kernel_launch(void* const* d_in, const int* in_sizes, int n_in,
                              void* d_out, int out_size, void* d_ws, size_t ws_size,
                              hipStream_t stream) {
  (void)in_sizes; (void)n_in; (void)out_size; (void)ws_size;

  const float* noisy_f = (const float*)d_in[0];
  const float* e1_w1 = (const float*)d_in[1];  const float* e1_b1 = (const float*)d_in[2];
  const float* e1_w2 = (const float*)d_in[3];  const float* e1_b2 = (const float*)d_in[4];
  const float* e2_w1 = (const float*)d_in[5];  const float* e2_b1 = (const float*)d_in[6];
  const float* e2_w2 = (const float*)d_in[7];  const float* e2_b2 = (const float*)d_in[8];
  const float* bn_w1 = (const float*)d_in[9];  const float* bn_b1 = (const float*)d_in[10];
  const float* bn_w2 = (const float*)d_in[11]; const float* bn_b2 = (const float*)d_in[12];
  const float* up2_w = (const float*)d_in[13]; const float* up2_b = (const float*)d_in[14];
  const float* dec2_w1 = (const float*)d_in[15]; const float* dec2_b1 = (const float*)d_in[16];
  const float* dec2_w2 = (const float*)d_in[17]; const float* dec2_b2 = (const float*)d_in[18];
  const float* up1_w = (const float*)d_in[19]; const float* up1_b = (const float*)d_in[20];
  const float* dec1_w1 = (const float*)d_in[21]; const float* dec1_b1 = (const float*)d_in[22];
  const float* dec1_w2 = (const float*)d_in[23]; const float* dec1_b2 = (const float*)d_in[24];
  const float* out_w = (const float*)d_in[25]; const float* out_b = (const float*)d_in[26];

  size_t off = 0;
  auto alloc = [&](size_t elems) -> f16* {
    f16* p = (f16*)d_ws + off;
    off += (elems + 127) & ~(size_t)127;
    return p;
  };

  // f16 weights (layer-1 OIHW; conv layers tap-major; tconv parity-packed)
  f16* w_e1_1 = alloc(64 * 27);
  f16* w_e1_2 = alloc(64 * 576);
  f16* w_e2_1 = alloc(128 * 576);   f16* w_e2_2 = alloc(128 * 1152);
  f16* w_bn_1 = alloc(256 * 1152);  f16* w_bn_2 = alloc(256 * 2304);
  f16* w_up2  = alloc(4 * 128 * 256);
  f16* w_d2_1 = alloc(128 * 2304);  f16* w_d2_2 = alloc(128 * 1152);
  f16* w_up1  = alloc(4 * 64 * 128);
  f16* w_d1_1 = alloc(64 * 1152);   f16* w_d1_2 = alloc(64 * 576);
  f16* w_out  = alloc(441 * 64);

  // f16 activations
  f16* noisy_h = alloc(3 * 65536);
  f16* x1   = alloc(64 * 65536);
  f16* p1   = alloc(64 * 16384);
  f16* x2   = alloc(128 * 16384);
  f16* p2   = alloc(128 * 4096);
  f16* bnk  = alloc(256 * 4096);
  f16* cat2 = alloc(256 * 16384);   // [u2 | x2 copy]
  f16* d2   = alloc(128 * 16384);
  f16* cat1 = alloc(128 * 65536);   // [u1 | x1 copy]
  f16* d1   = alloc(64 * 65536);
  f16* tmpa = alloc(64 * 65536);
  f16* tmpb = alloc(128 * 16384);
  f16* tmpc = alloc(256 * 4096);

  auto cvt = [&](const float* s, f16* d, int n) {
    cvt_f32_f16_kernel<<<gs_blocks(n), 256, 0, stream>>>(s, d, n);
  };
  auto repack = [&](const float* s, f16* d, int Cin, int Cout) {
    repack_conv_tapmajor<<<gs_blocks((long)Cout * 9 * Cin), 256, 0, stream>>>(s, d, Cin, Cout);
  };

  cvt(noisy_f, noisy_h, 3 * 65536);
  cvt(e1_w1, w_e1_1, 64 * 27);                    // layer-1 stays OIHW
  repack(e1_w2, w_e1_2, 64, 64);
  repack(e2_w1, w_e2_1, 64, 128);  repack(e2_w2, w_e2_2, 128, 128);
  repack(bn_w1, w_bn_1, 128, 256); repack(bn_w2, w_bn_2, 256, 256);
  repack(dec2_w1, w_d2_1, 256, 128); repack(dec2_w2, w_d2_2, 128, 128);
  repack(dec1_w1, w_d1_1, 128, 64);  repack(dec1_w2, w_d1_2, 64, 64);
  cvt(out_w, w_out, 441 * 64);
  repack_tconv_kernel<<<gs_blocks(4 * 128 * 256), 256, 0, stream>>>(up2_w, w_up2, 256, 128);
  repack_tconv_kernel<<<gs_blocks(4 * 64 * 128), 256, 0, stream>>>(up1_w, w_up1, 128, 64);

  // Tiled conv: one block per 16-pixel strip; LDS = Cin*54 halfs.
  auto conv = [&](const f16* x, const f16* w, const float* b, f16* y,
                  int Cin, int cinShift, int Cout, int H, int W) {
    const int nTilesN = (H * W) >> 4;
    const size_t shmem = (size_t)Cin * 54 * sizeof(f16);
    conv3_tiled_wmma<<<nTilesN, 256, shmem, stream>>>(x, w, b, y, Cin, cinShift, Cout, H, W, 1);
  };
  auto tconv = [&](const f16* x, const f16* wt, const float* b, f16* y,
                   int Cin, int Cout, int Hin, int Win) {
    const int nTilesN = (Hin * Win) >> 4;
    const size_t shmem = (size_t)16 * Cin * sizeof(f16);
    tconv_tiled_wmma<<<nTilesN, 256, shmem, stream>>>(x, wt, b, y, Cin, Cout, Hin, Win);
  };

  // ---- UNet pipeline ----
  {   // layer 1 (Cin=3): generic guarded kernel, 8 waves/block
    const int nTilesN = 65536 >> 4;
    const int tiles = 4 * nTilesN;
    conv_wmma_generic<<<(tiles + 7) / 8, 256, 0, stream>>>(
        noisy_h, w_e1_1, e1_b1, tmpa, 3, 64, 256, 256, 1, nTilesN);
  }
  conv(tmpa, w_e1_2, e1_b2, x1, 64, 6, 64, 256, 256);
  maxpool2_kernel<<<gs_blocks(64 * 16384), 256, 0, stream>>>(x1, p1, 64, 256, 256);

  conv(p1, w_e2_1, e2_b1, tmpb, 64, 6, 128, 128, 128);
  conv(tmpb, w_e2_2, e2_b2, x2, 128, 7, 128, 128, 128);
  maxpool2_kernel<<<gs_blocks(128 * 4096), 256, 0, stream>>>(x2, p2, 128, 128, 128);

  conv(p2, w_bn_1, bn_b1, tmpc, 128, 7, 256, 64, 64);
  conv(tmpc, w_bn_2, bn_b2, bnk, 256, 8, 256, 64, 64);

  tconv(bnk, w_up2, up2_b, cat2, 256, 128, 64, 64);                 // u2 -> cat2[0:128]
  copy_f16_kernel<<<gs_blocks(128 * 16384), 256, 0, stream>>>(
      x2, cat2 + (size_t)128 * 16384, 128 * 16384);
  conv(cat2, w_d2_1, dec2_b1, tmpb, 256, 8, 128, 128, 128);
  conv(tmpb, w_d2_2, dec2_b2, d2, 128, 7, 128, 128, 128);

  tconv(d2, w_up1, up1_b, cat1, 128, 64, 128, 128);                 // u1 -> cat1[0:64]
  copy_f16_kernel<<<gs_blocks(64 * 65536), 256, 0, stream>>>(
      x1, cat1 + (size_t)64 * 65536, 64 * 65536);
  conv(cat1, w_d1_1, dec1_b1, tmpa, 128, 7, 64, 256, 256);
  conv(tmpa, w_d1_2, dec1_b2, d1, 64, 6, 64, 256, 256);

  // ---- Fused KPN head: 4096 tiles of 16 pixels, one wave each ----
  kpn_wmma_kernel<<<4096, 32, 0, stream>>>(d1, w_out, out_b, noisy_f, (float*)d_out);
}
